// Int4Linear_12936441496158
// MI455X (gfx1250) — 8-bit matrix operand order repaired and checked by reading, compiled, not run on hardware
//
#include <hip/hip_runtime.h>

typedef __attribute__((ext_vector_type(8))) int v8i;
typedef __attribute__((ext_vector_type(4))) int v4i;
typedef __attribute__((ext_vector_type(2))) int v2i;

typedef __attribute__((ext_vector_type(4))) unsigned int u32x4;
typedef __attribute__((ext_vector_type(8))) int i32x8;
typedef __attribute__((ext_vector_type(4))) int i32x4;

#if defined(__AMDGCN__) && __has_builtin(__builtin_amdgcn_tensor_load_to_lds) && \
    __has_builtin(__builtin_amdgcn_s_wait_tensorcnt)
#define USE_TDM 1
#else
#define USE_TDM 0
#endif

__global__ __launch_bounds__(256, 1)
void quant_rows_kernel(const _Float16* __restrict__ in,
                       signed char* __restrict__ q,
                       float* __restrict__ scales, int K)
{
    __shared__ float red[256];
    const int row = blockIdx.x;
    const int tid = threadIdx.x;
    const size_t base = (size_t)row * K + (size_t)tid * 16;

    _Float16 v[16] __attribute__((aligned(16)));
    *(int4*)(&v[0]) = *(const int4*)(in + base);
    *(int4*)(&v[8]) = *(const int4*)(in + base + 8);

    float m = 0.0f;
#pragma unroll
    for (int i = 0; i < 16; ++i) m = fmaxf(m, fabsf((float)v[i]));
    red[tid] = m;
    __syncthreads();
#pragma unroll
    for (int s = 128; s > 0; s >>= 1) {
        if (tid < s) red[tid] = fmaxf(red[tid], red[tid + s]);
        __syncthreads();
    }
    const float scale = fmaxf(red[0] * (1.0f / 7.0f), 1e-8f);
    if (tid == 0) scales[row] = scale;
    const float inv = 1.0f / scale;

    signed char o[16] __attribute__((aligned(16)));
#pragma unroll
    for (int i = 0; i < 16; ++i) {
        float t = rintf((float)v[i] * inv);
        t = fminf(fmaxf(t, -8.0f), 7.0f);
        o[i] = (signed char)t;
    }
    *(int4*)(q + base) = *(const int4*)o;
}

#if USE_TDM
__device__ __forceinline__ void tdm_load_tile_128x128(const void* gsrc,
                                                      unsigned lds_off,
                                                      int rowstride  )
{
    const unsigned long long ga = (unsigned long long)(uintptr_t)gsrc;
    u32x4 g0;
    g0[0] = 1u;
    g0[1] = lds_off;
    g0[2] = (unsigned)(ga & 0xffffffffu);
    g0[3] = (unsigned)((ga >> 32) & 0x1ffffffu)
          | (2u << 30);

    i32x8 g1;
    g1[0] = (1 << 20)
          | (4 << 22)
          | (3 << 25);
    g1[1] = (rowstride & 0xffff) << 16;
    g1[2] = ((rowstride >> 16) & 0xffff) | (128 << 16);
    g1[3] = (128 << 16);
    g1[4] = 128;
    g1[5] = rowstride;
    g1[6] = 0;
    g1[7] = 0;

    const i32x4 gz = {0, 0, 0, 0};
#if __clang_major__ >= 23
    const i32x8 gz8 = {0, 0, 0, 0, 0, 0, 0, 0};
    __builtin_amdgcn_tensor_load_to_lds(g0, g1, gz, gz, gz8, 0);
#else
    __builtin_amdgcn_tensor_load_to_lds(g0, g1, gz, gz, 0);
#endif
}
#endif

__device__ __forceinline__ v8i frag_a(const char* sm, int lane, int mrow, int kk)
{
    const char* rp = sm + (size_t)mrow * 144;
    const int kb = kk + ((lane & 16) ? 16 : 0);
    v2i t0 = *(const v2i*)(rp + kb);
    v2i t1 = *(const v2i*)(rp + kb + 8);
    v2i t2 = *(const v2i*)(rp + kb + 32);
    v2i t3 = *(const v2i*)(rp + kb + 40);
    v8i a = { t0[0], t0[1], t1[0], t1[1], t2[0], t2[1], t3[0], t3[1] };
    return a;
}

__device__ __forceinline__ v8i frag_b(const char* sm, int lane, int nrow, int kk)
{
    const char* rp = sm + (size_t)nrow * 144;
    const int kb0 = kk + ((lane & 16) ? 16 : 0);
    const int kb1 = kk + ((lane & 16) ? 48 : 32);
    v4i u0 = *(const v4i*)(rp + kb0);
    v4i u1 = *(const v4i*)(rp + kb1);
    v8i b = { u0[0], u0[1], u0[2], u0[3], u1[0], u1[1], u1[2], u1[3] };
    return b;
}

__global__ __launch_bounds__(256, 1)
void int4_gemm_wmma(const signed char* __restrict__ xq,
                    const signed char* __restrict__ wq,
                    const float* __restrict__ xs,
                    const float* __restrict__ wsc,
                    const float* __restrict__ bias,
                    float* __restrict__ out,
                    int M, int N, int K)
{
    __shared__ int4 smA[2][128][9];
    __shared__ int4 smB[2][128][9];

    const int tid  = threadIdx.x;
    const int lane = tid & 31;
    const int wave = tid >> 5;
    const int wm = wave & 3;
    const int wn = wave >> 2;
    const int bM = blockIdx.y * 128;
    const int bN = blockIdx.x * 128;
    const int ntiles = K >> 7;

    const v8i vzero = {0, 0, 0, 0, 0, 0, 0, 0};
    v8i acc[2][4];
#pragma unroll
    for (int i = 0; i < 2; ++i)
#pragma unroll
        for (int j = 0; j < 4; ++j) acc[i][j] = vzero;

    const signed char* gA = xq + (size_t)bM * K;
    const signed char* gB = wq + (size_t)bN * K;

#if USE_TDM
    if (wave == 0) {
        tdm_load_tile_128x128(gA, (unsigned)(uintptr_t)&smA[0][0][0], K);
        tdm_load_tile_128x128(gB, (unsigned)(uintptr_t)&smB[0][0][0], K);
        if (ntiles > 1) {
            tdm_load_tile_128x128(gA + 128, (unsigned)(uintptr_t)&smA[1][0][0], K);
            tdm_load_tile_128x128(gB + 128, (unsigned)(uintptr_t)&smB[1][0][0], K);
            __builtin_amdgcn_s_wait_tensorcnt(2);
        } else {
            __builtin_amdgcn_s_wait_tensorcnt(0);
        }
    }
    __syncthreads();

    for (int t = 0; t < ntiles; ++t) {
        const int buf = t & 1;
        const char* baseA = (const char*)smA[buf];
        const char* baseB = (const char*)smB[buf];
#pragma unroll
        for (int kk = 0; kk < 128; kk += 64) {
            v8i a[2];
#pragma unroll
            for (int i = 0; i < 2; ++i)
                a[i] = frag_a(baseA, lane, wm * 32 + i * 16 + (lane & 15), kk);
            v8i b[4];
#pragma unroll
            for (int j = 0; j < 4; ++j)
                b[j] = frag_b(baseB, lane, wn * 64 + j * 16 + (lane & 15), kk);
#pragma unroll
            for (int i = 0; i < 2; ++i)
#pragma unroll
                for (int j = 0; j < 4; ++j)
                    acc[i][j] = __builtin_amdgcn_wmma_i32_16x16x64_iu8(
                        true, a[i], true, b[j], acc[i][j], false, false);
        }
        __syncthreads();
        if (wave == 0) {
            if (t + 2 < ntiles) {
                const int k2 = (t + 2) << 7;
                tdm_load_tile_128x128(gA + k2, (unsigned)(uintptr_t)&smA[buf][0][0], K);
                tdm_load_tile_128x128(gB + k2, (unsigned)(uintptr_t)&smB[buf][0][0], K);
                __builtin_amdgcn_s_wait_tensorcnt(2);
            } else if (t + 1 < ntiles) {
                __builtin_amdgcn_s_wait_tensorcnt(0);
            }
        }
        __syncthreads();
    }
#else
    for (int k0 = 0; k0 < K; k0 += 128) {
        __syncthreads();
#pragma unroll
        for (int it = 0; it < 4; ++it) {
            const int idx = tid + it * 256;
            const int row = idx >> 3;
            const int c   = idx & 7;
            smA[0][row][c] = *(const int4*)(gA + (size_t)row * K + k0 + c * 16);
            smB[0][row][c] = *(const int4*)(gB + (size_t)row * K + k0 + c * 16);
            if (c == 0 && (k0 + 128) < K) {
                __builtin_prefetch(gA + (size_t)row * K + k0 + 128, 0, 0);
                __builtin_prefetch(gB + (size_t)row * K + k0 + 128, 0, 0);
            }
        }
        __syncthreads();
#pragma unroll
        for (int kk = 0; kk < 128; kk += 64) {
            v8i a[2];
#pragma unroll
            for (int i = 0; i < 2; ++i)
                a[i] = frag_a((const char*)smA[0], lane, wm * 32 + i * 16 + (lane & 15), kk);
            v8i b[4];
#pragma unroll
            for (int j = 0; j < 4; ++j)
                b[j] = frag_b((const char*)smB[0], lane, wn * 64 + j * 16 + (lane & 15), kk);
#pragma unroll
            for (int i = 0; i < 2; ++i)
#pragma unroll
                for (int j = 0; j < 4; ++j)
                    acc[i][j] = __builtin_amdgcn_wmma_i32_16x16x64_iu8(
                        true, a[i], true, b[j], acc[i][j], false, false);
        }
    }
#endif

    float sx[2][8];
#pragma unroll
    for (int i = 0; i < 2; ++i) {
        const int rb = bM + wm * 32 + i * 16 + ((lane & 16) ? 8 : 0);
#pragma unroll
        for (int v = 0; v < 8; ++v) sx[i][v] = xs[rb + v];
    }
#pragma unroll
    for (int j = 0; j < 4; ++j) {
        const int c  = bN + wn * 64 + j * 16 + (lane & 15);
        const float sw = wsc[c];
        const float bc = bias[c];
#pragma unroll
        for (int i = 0; i < 2; ++i) {
            const int rb = bM + wm * 32 + i * 16 + ((lane & 16) ? 8 : 0);
#pragma unroll
            for (int v = 0; v < 8; ++v) {
                const float val = (float)acc[i][j][v] * sx[i][v] * sw + bc;
                __builtin_nontemporal_store(val, &out[(size_t)(rb + v) * N + c]);
            }
        }
    }
}

extern "C" void kernel_launch(void* const* d_in, const int* in_sizes, int n_in,
                              void* d_out, int out_size, void* d_ws, size_t ws_size,
                              hipStream_t stream)
{
    const _Float16* x    = (const _Float16*)d_in[0];
    const _Float16* w    = (const _Float16*)d_in[1];
    const float*    bias = (const float*)d_in[2];
    float*          out  = (float*)d_out;

    const int N = in_sizes[2];
    const int K = in_sizes[1] / N;
    const int M = in_sizes[0] / K;

    signed char* xq = (signed char*)d_ws;
    signed char* wq = xq + (size_t)M * K;
    float* xs = (float*)(wq + (size_t)N * K);
    float* ws = xs + M;

    quant_rows_kernel<<<M, 256, 0, stream>>>(x, xq, xs, K);
    quant_rows_kernel<<<N, 256, 0, stream>>>(w, wq, ws, K);

    dim3 grid(N / 128, M / 128);
    int4_gemm_wmma<<<grid, 256, 0, stream>>>(xq, wq, xs, ws, bias, out, M, N, K);
}
